// GIN_AC_44135083934291
// MI455X (gfx1250) — matrix operand order repaired and checked by reading, not compiled, not run on hardware
//
#include <hip/hip_runtime.h>
#include <hip/hip_bf16.h>

// ---------------------------------------------------------------------------
// Problem constants (from the reference)
// ---------------------------------------------------------------------------
#define NNODE 50000
#define MPAD  50016              // NNODE rounded up to multiple of 32
#define NEDGE 800000
#define NGRAPH 512
#define NODE_DIM 52
#define CONV 256
#define NLAYER 4

typedef __attribute__((ext_vector_type(16))) __bf16 v16bf;
typedef __attribute__((ext_vector_type(8)))  __bf16 v8bf;
typedef __attribute__((ext_vector_type(8)))  float  v8f;

static __device__ __forceinline__ unsigned bfbits(float x) {
    // round-to-nearest-even fp32 -> bf16 raw bits (low 16)
    unsigned u = __builtin_bit_cast(unsigned, x);
    unsigned r = u + 0x7FFFu + ((u >> 16) & 1u);
    return r >> 16;
}

static __device__ __forceinline__ __bf16 f2bf(float x) {
    unsigned short h = (unsigned short)bfbits(x);
    return __builtin_bit_cast(__bf16, h);
}

static __device__ __forceinline__ void atomicMaxF(float* addr, float val) {
    // monotone int/uint mapping trick; addr initialized to -inf
    if (val >= 0.0f)
        atomicMax((int*)addr, __float_as_int(val));
    else
        atomicMin((unsigned int*)addr, (unsigned int)__float_as_int(val));
}

// ---------------------------------------------------------------------------
// Utility fills
// ---------------------------------------------------------------------------
__global__ void fill_f32(float* __restrict__ p, long n, float v) {
    long i = (long)blockIdx.x * blockDim.x + threadIdx.x;
    if (i < n) p[i] = v;
}

// out[n][f] = (f < in_d) ? (1+eps)*in[n][f] : 0   (row stride in_d -> outs)
__global__ void scale_copy_pad(float* __restrict__ out, const float* __restrict__ in,
                               const float* __restrict__ eps, int N, int in_d, int outs) {
    long i = (long)blockIdx.x * blockDim.x + threadIdx.x;
    if (i >= (long)N * outs) return;
    int n = (int)(i / outs);
    int f = (int)(i % outs);
    float v = 0.0f;
    if (f < in_d) v = (1.0f + eps[0]) * in[(long)n * in_d + f];
    out[i] = v;
}

// ---------------------------------------------------------------------------
// Weight prep: WT[n][k] = bf16(W[k][n]), K zero-padded to Kpad (mult. of 32).
// Makes the WMMA B-fragment a contiguous 32-byte load with no bounds checks.
// ---------------------------------------------------------------------------
__global__ void wprep(__bf16* __restrict__ wt, const float* __restrict__ W,
                      int K, int Kpad) {
    long i = (long)blockIdx.x * blockDim.x + threadIdx.x;
    if (i >= (long)CONV * Kpad) return;
    int n = (int)(i / Kpad);
    int k = (int)(i % Kpad);
    wt[i] = f2bf((k < K) ? W[(long)k * CONV + n] : 0.0f);
}

// ---------------------------------------------------------------------------
// Fused GINE edge kernel: msg = relu(h[src] + edge_attr @ We + be)
// scatter-add into agg[dst] (row stride `as`; agg pre-init to (1+eps)*h).
// one thread per (edge, 4 features)
// ---------------------------------------------------------------------------
__global__ void edge_gine(float* __restrict__ agg, const float* __restrict__ h,
                          const float* __restrict__ ea, const float* __restrict__ We,
                          const float* __restrict__ be, const int* __restrict__ src,
                          const int* __restrict__ dst, long total4, int in_d, int as) {
    long idx = (long)blockIdx.x * blockDim.x + threadIdx.x;
    if (idx >= total4) return;
    const int qpe = in_d >> 2;              // feature-quads per edge
    int e = (int)(idx / qpe);
    int f = ((int)(idx % qpe)) << 2;
    int s = src[e], d = dst[e];

    float ev[10];
#pragma unroll
    for (int k = 0; k < 10; ++k) ev[k] = ea[(long)e * 10 + k];   // broadcast loads

    float4 acc = *(const float4*)(be + f);
#pragma unroll
    for (int k = 0; k < 10; ++k) {
        const float4 w = *(const float4*)(We + (long)k * in_d + f);
        acc.x += ev[k] * w.x; acc.y += ev[k] * w.y;
        acc.z += ev[k] * w.z; acc.w += ev[k] * w.w;
    }
    const float4 hv = *(const float4*)(h + (long)s * in_d + f);
    float* ap = agg + (long)d * as + f;
    atomicAdd(ap + 0, fmaxf(hv.x + acc.x, 0.0f));
    atomicAdd(ap + 1, fmaxf(hv.y + acc.y, 0.0f));
    atomicAdd(ap + 2, fmaxf(hv.z + acc.z, 0.0f));
    atomicAdd(ap + 3, fmaxf(hv.w + acc.w, 0.0f));
}

// nagg[dst] += h[src]  (CONV features, 4 per thread)
__global__ void edge_nagg(float* __restrict__ nagg, const float* __restrict__ h,
                          const int* __restrict__ src, const int* __restrict__ dst,
                          long total4) {
    long i = (long)blockIdx.x * blockDim.x + threadIdx.x;
    if (i >= total4) return;
    int e = (int)(i >> 6);                  // 64 quads per edge
    int f = ((int)(i & 63)) << 2;
    const float4 hv = *(const float4*)(h + (long)src[e] * CONV + f);
    float* np = nagg + (long)dst[e] * CONV + f;
    atomicAdd(np + 0, hv.x);
    atomicAdd(np + 1, hv.y);
    atomicAdd(np + 2, hv.z);
    atomicAdd(np + 3, hv.w);
}

// ---------------------------------------------------------------------------
// WMMA GEMM: C[M,256] = act(A[*,Kpad] @ W[Kpad,256] + bias)
// A row stride == Kpad, rows readable up to gridDim*32 (padded buffers);
// W pre-transposed/padded to bf16 WT[256][Kpad].
// block = 512 threads = 16 waves; block owns 32 rows x 256 cols.
// Wave w owns columns [16w,16w+16), two 16-row tiles (2 WMMA / k-step, shared
// B fragment). Double-buffered LDS A-tile, branch-free software pipeline.
// ---------------------------------------------------------------------------
__global__ void __launch_bounds__(512)
wmma_gemm_n256(const float* __restrict__ A, const __bf16* __restrict__ WT,
               const float* __restrict__ bias, float* __restrict__ C,
               int M, int Kpad, int do_relu) {
    __shared__ __bf16 lds_a[2][32 * 32];
    const int tid  = threadIdx.x;
    const int wave = tid >> 5;          // 0..15 -> N tile
    const int lane = tid & 31;
    const int half = lane >> 4;         // 0 / 1
    const int mloc = lane & 15;         // row within 16-row tile (A frag)
    const int m0   = blockIdx.x * 32;
    const int ncol = wave * 16 + mloc;  // output column for this lane (B/C)

    // staging coords: thread fills 2 consecutive k of one row per k-step
    const int sr  = tid >> 4;           // 0..31 (row in 32-row tile)
    const int sk2 = (tid & 15) << 1;    // 0,2,...,30

    const float*  ap  = A  + (long)(m0 + sr) * Kpad + sk2;
    const __bf16* wtp = WT + (long)ncol * Kpad + 16 * half;

    v8f acc0 = {0.f, 0.f, 0.f, 0.f, 0.f, 0.f, 0.f, 0.f};
    v8f acc1 = acc0;
    const int ksteps = Kpad >> 5;

    // ---- prologue: load k-step 0 ----
    float2 ar  = *(const float2*)(ap);
    v16bf bcur = *(const v16bf*)(wtp);
    v16bf bnxt;

    int p = 0;
    for (int kt = 0; kt < ksteps; ++kt) {
        // pack two bf16 -> one ds_store_b32
        unsigned w = bfbits(ar.x) | (bfbits(ar.y) << 16);
        *(unsigned*)(&lds_a[p][sr * 32 + sk2]) = w;
        __syncthreads();

        // branch-free prefetch of next k-step (wraps to 0 on last iteration)
        const int kn = (kt + 1 < ksteps) ? ((kt + 1) << 5) : 0;
        ar   = *(const float2*)(ap + kn);
        bnxt = *(const v16bf*)(wtp + kn);

        // A fragments: two contiguous 16B runs per lane, per 16-row tile
        const __bf16* lp0 = &lds_a[p][mloc * 32];
        const __bf16* lp1 = &lds_a[p][(16 + mloc) * 32];
        const v8bf a0lo = *(const v8bf*)(lp0 + 16 * half);
        const v8bf a0hi = *(const v8bf*)(lp0 + 16 * half + 8);
        const v8bf a1lo = *(const v8bf*)(lp1 + 16 * half);
        const v8bf a1hi = *(const v8bf*)(lp1 + 16 * half + 8);
        v16bf a0, a1;
#pragma unroll
        for (int e = 0; e < 8; ++e) {
            a0[e] = a0lo[e]; a0[e + 8] = a0hi[e];
            a1[e] = a1lo[e]; a1[e + 8] = a1hi[e];
        }

        acc0 = __builtin_amdgcn_wmma_f32_16x16x32_bf16(
                   false, a0, false, bcur, (short)0, acc0, false, false);
        acc1 = __builtin_amdgcn_wmma_f32_16x16x32_bf16(
                   false, a1, false, bcur, (short)0, acc1, false, false);
        bcur = bnxt;
        p ^= 1;
    }

    const float bv = bias[ncol];
#pragma unroll
    for (int r = 0; r < 8; ++r) {
        int m = m0 + r + 8 * half;      // tile0 row
        if (m < M) {
            float v = acc0[r] + bv;
            if (do_relu) v = fmaxf(v, 0.0f);
            C[(long)m * CONV + ncol] = v;
        }
        int m2 = m + 16;                // tile1 row
        if (m2 < M) {
            float v = acc1[r] + bv;
            if (do_relu) v = fmaxf(v, 0.0f);
            C[(long)m2 * CONV + ncol] = v;
        }
    }
}

// ---------------------------------------------------------------------------
// BatchNorm (training-mode batch stats over N rows, 256 channels)
// ---------------------------------------------------------------------------
__global__ void bn_stats(const float* __restrict__ z, float* __restrict__ stats, int N) {
    int f = threadIdx.x;                 // channel
    float s = 0.f, sq = 0.f;
    for (int n = blockIdx.x; n < N; n += gridDim.x) {
        float v = z[(long)n * CONV + f];
        s += v; sq += v * v;
    }
    atomicAdd(&stats[f], s);
    atomicAdd(&stats[CONV + f], sq);
}

__global__ void bn_apply(float* __restrict__ h, const float* __restrict__ z,
                         const float* __restrict__ stats,
                         const float* __restrict__ gamma,
                         const float* __restrict__ beta, int N) {
    long i = (long)blockIdx.x * blockDim.x + threadIdx.x;
    if (i >= (long)N * CONV) return;
    int f = (int)(i & 255);
    float invn = 1.0f / (float)N;
    float mu  = stats[f] * invn;
    float var = stats[CONV + f] * invn - mu * mu;
    h[i] = (z[i] - mu) * rsqrtf(var + 1e-5f) * gamma[f] + beta[f];
}

// ---------------------------------------------------------------------------
// Attention pool: logits = nagg@Wrel + brel + h@Wroot ; warp per node,
// fused atomic-max into per-graph max buffer.
// ---------------------------------------------------------------------------
__global__ void attn_logits(float* __restrict__ logits, const float* __restrict__ nagg,
                            const float* __restrict__ h, const float* __restrict__ Wrel,
                            const float* __restrict__ Wroot, const float* __restrict__ brel,
                            const int* __restrict__ batch, float* __restrict__ mbuf, int N) {
    int warp = threadIdx.x >> 5;
    int lane = threadIdx.x & 31;
    int n = blockIdx.x * 8 + warp;
    if (n >= N) return;
    float p = 0.f;
#pragma unroll
    for (int j = lane; j < CONV; j += 32)
        p += nagg[(long)n * CONV + j] * Wrel[j] + h[(long)n * CONV + j] * Wroot[j];
#pragma unroll
    for (int off = 16; off > 0; off >>= 1)
        p += __shfl_down(p, off, 32);
    if (lane == 0) {
        float lg = p + brel[0];
        logits[n] = lg;
        atomicMaxF(&mbuf[batch[n]], lg);
    }
}

__global__ void attn_exp_den(float* __restrict__ ex, const float* __restrict__ logits,
                             const float* __restrict__ mbuf, float* __restrict__ den,
                             const int* __restrict__ batch, int N) {
    int n = blockIdx.x * blockDim.x + threadIdx.x;
    if (n >= N) return;
    float e = expf(logits[n] - mbuf[batch[n]]);
    ex[n] = e;
    atomicAdd(&den[batch[n]], e);
}

__global__ void attn_pool(float* __restrict__ pooled, const float* __restrict__ h,
                          const float* __restrict__ ex, const float* __restrict__ den,
                          const int* __restrict__ batch, int layer, int N) {
    long i = (long)blockIdx.x * blockDim.x + threadIdx.x;
    if (i >= (long)N * CONV) return;
    int n = (int)(i >> 8);
    int f = (int)(i & 255);
    int g = batch[n];
    float s = ex[n] / (den[g] + 1e-16f);
    atomicAdd(&pooled[(long)g * (CONV * NLAYER) + layer * CONV + f], h[i] * s);
}

// ---------------------------------------------------------------------------
// Host-side launch
// ---------------------------------------------------------------------------
static inline unsigned gb(long total, int threads) {
    return (unsigned)((total + threads - 1) / threads);
}

extern "C" void kernel_launch(void* const* d_in, const int* in_sizes, int n_in,
                              void* d_out, int out_size, void* d_ws, size_t ws_size,
                              hipStream_t stream) {
    (void)in_sizes; (void)n_in; (void)out_size; (void)ws_size;
    const int N = NNODE, E = NEDGE, G = NGRAPH;

    // ---- input pointer map (setup dict order; params flattened sorted-key) ----
    const float* x  = (const float*)d_in[0];                    // [N,52]
    const float* ea = (const float*)d_in[1];                    // [E,10]
    // d_in[2] = prot_feat (unused)
    // layers: base 3 + 9*l : W1,W2,We,b1,b2,be,beta,eps,gamma
    const float* oW1 = (const float*)d_in[39];                  // [1024,256]
    const float* oW2 = (const float*)d_in[40];                  // [256,256]
    const float* ob1 = (const float*)d_in[41];
    const float* ob2 = (const float*)d_in[42];
    const float* Wrel  = (const float*)d_in[43];                // [256]
    const float* Wroot = (const float*)d_in[44];                // [256]
    const float* brel  = (const float*)d_in[45];                // [1]
    const int* ei    = (const int*)d_in[46];                    // [2,E]
    const int* src   = ei;
    const int* dstp  = ei + E;
    const int* batch = (const int*)d_in[47];                    // [N]

    // ---- workspace carve (node buffers padded to MPAD rows for the GEMM) ----
    float* ws = (float*)d_ws;
    const long NF  = (long)NNODE * CONV;
    const long NFP = (long)MPAD  * CONV;
    float* h      = ws;                 ws += NFP;      // node features (post-BN)
    float* agg    = ws;                 ws += NFP;      // (1+eps)h + msg sum / z2
    float* z1     = ws;                 ws += NFP;      // hidden of layer MLP
    float* nagg   = ws;                 ws += NF;       // neighbor sum for pool
    float* logits = ws;                 ws += N;
    float* exv    = ws;                 ws += N;
    float* mbuf   = ws;                 ws += G;
    float* den    = ws;                 ws += G;
    float* stats  = ws;                 ws += 2 * CONV;
    float* pooled = ws;                 ws += (long)G * CONV * NLAYER; // [G,1024]
    float* gtmp   = ws;                 ws += (long)G * CONV;          // [G,256]
    __bf16* wt    = (__bf16*)ws;        // 256*1024 bf16 max (32B-aligned carve)
    float* out    = (float*)d_out;      // [G,256]

    fill_f32<<<gb((long)G * CONV * NLAYER, 256), 256, 0, stream>>>(
        pooled, (long)G * CONV * NLAYER, 0.0f);

    for (int l = 0; l < NLAYER; ++l) {
        const int base = 3 + 9 * l;
        const float* W1    = (const float*)d_in[base + 0];
        const float* W2    = (const float*)d_in[base + 1];
        const float* We    = (const float*)d_in[base + 2];
        const float* b1    = (const float*)d_in[base + 3];
        const float* b2    = (const float*)d_in[base + 4];
        const float* be    = (const float*)d_in[base + 5];
        const float* beta  = (const float*)d_in[base + 6];
        const float* eps   = (const float*)d_in[base + 7];
        const float* gamma = (const float*)d_in[base + 8];

        const float* inH = (l == 0) ? x : h;
        const int in_d = (l == 0) ? NODE_DIM : CONV;    // logical input dim
        const int kpad = (l == 0) ? 64 : CONV;          // padded dim == agg stride

        // agg[n][0:kpad] = (1+eps)*h | 0-pad ; then agg[dst] += relu(h[src]+lin_e)
        scale_copy_pad<<<gb((long)N * kpad, 256), 256, 0, stream>>>(
            agg, inH, eps, N, in_d, kpad);
        edge_gine<<<gb((long)E * (in_d >> 2), 256), 256, 0, stream>>>(
            agg, inH, ea, We, be, src, dstp, (long)E * (in_d >> 2), in_d, kpad);

        // z1 = relu(agg @ W1 + b1) ; agg(z2) = relu(z1 @ W2 + b2)
        wprep<<<gb((long)CONV * kpad, 256), 256, 0, stream>>>(wt, W1, in_d, kpad);
        wmma_gemm_n256<<<gb(N, 32), 512, 0, stream>>>(agg, wt, b1, z1, N, kpad, 1);
        wprep<<<gb((long)CONV * CONV, 256), 256, 0, stream>>>(wt, W2, CONV, CONV);
        wmma_gemm_n256<<<gb(N, 32), 512, 0, stream>>>(z1, wt, b2, agg, N, CONV, 1);

        // BatchNorm (batch statistics) -> h
        fill_f32<<<gb(2 * CONV, 256), 256, 0, stream>>>(stats, 2 * CONV, 0.0f);
        bn_stats<<<256, 256, 0, stream>>>(agg, stats, N);
        bn_apply<<<gb(NF, 256), 256, 0, stream>>>(h, agg, stats, gamma, beta, N);

        // Global attention pool
        fill_f32<<<gb(NF, 256), 256, 0, stream>>>(nagg, NF, 0.0f);
        edge_nagg<<<gb((long)E * 64, 256), 256, 0, stream>>>(nagg, h, src, dstp,
                                                             (long)E * 64);
        fill_f32<<<gb(G, 256), 256, 0, stream>>>(mbuf, G, -__builtin_inff());
        attn_logits<<<gb(N, 8), 256, 0, stream>>>(logits, nagg, h, Wrel, Wroot,
                                                  brel, batch, mbuf, N);
        fill_f32<<<gb(G, 256), 256, 0, stream>>>(den, G, 0.0f);
        attn_exp_den<<<gb(N, 256), 256, 0, stream>>>(exv, logits, mbuf, den, batch, N);
        attn_pool<<<gb(NF, 256), 256, 0, stream>>>(pooled, h, exv, den, batch, l, N);
    }

    // Output MLP: out = relu(pooled @ oW1 + ob1) @ oW2 + ob2
    wprep<<<gb((long)CONV * (CONV * NLAYER), 256), 256, 0, stream>>>(
        wt, oW1, CONV * NLAYER, CONV * NLAYER);
    wmma_gemm_n256<<<gb(G, 32), 512, 0, stream>>>(pooled, wt, ob1, gtmp, G,
                                                  CONV * NLAYER, 1);
    wprep<<<gb((long)CONV * CONV, 256), 256, 0, stream>>>(wt, oW2, CONV, CONV);
    wmma_gemm_n256<<<gb(G, 32), 512, 0, stream>>>(gtmp, wt, ob2, out, G, CONV, 0);
}
